// MultiHeadAttention_24481313587874
// MI455X (gfx1250) — matrix operand order repaired and checked by reading, not compiled, not run on hardware
//
#include <hip/hip_runtime.h>

#define NB 4
#define NS 2048
#define NC 1024
#define NH 16
#define NHS 64

typedef __bf16 bf16;
typedef __attribute__((ext_vector_type(16))) __bf16 v16bf;
typedef __attribute__((ext_vector_type(8)))  __bf16 v8bf;
typedef __attribute__((ext_vector_type(8)))  float  v8f;
typedef __attribute__((ext_vector_type(4)))  int    v4i;

// ---- optional CDNA5 async global->LDS path (compile-safe via __has_builtin) ----
#ifdef __HIP_DEVICE_COMPILE__
#ifdef __has_builtin
#if __has_builtin(__builtin_amdgcn_global_load_async_to_lds_b128) && \
    __has_builtin(__builtin_amdgcn_s_wait_asynccnt)
#define USE_ASYNC_LDS 1
#endif
#endif
#endif
#ifndef USE_ASYNC_LDS
#define USE_ASYNC_LDS 0
#endif

#if USE_ASYNC_LDS
typedef __attribute__((address_space(1))) v4i GV4;   // global int4
typedef __attribute__((address_space(3))) v4i LV4;   // LDS int4
static __device__ __forceinline__ GV4* as_global_v4(const void* p) {
  return (GV4*)(unsigned long long)p;                // global generic == as1 bits
}
static __device__ __forceinline__ LV4* as_lds_v4(const void* p) {
  // ISA aperture rule: the LDS address is the low 32 bits of the generic shared pointer.
  return (LV4*)(unsigned int)(unsigned long long)p;
}
#endif

static __device__ __forceinline__ v8f wmma_bf16(v16bf a, v16bf b, v8f c) {
  // (neg_a, A, neg_b, B, c_mod, C, reuse_a, reuse_b)
  return __builtin_amdgcn_wmma_f32_16x16x32_bf16(false, a, false, b, (short)0, c,
                                                 false, false);
}

// A fragment (16x32 bf16, M x K) from row-major memory, row stride rs (elems).
// lane: m = lane&15 (row M), hi = lane>>4.
// element j (0..15): K = hi*16+j.
static __device__ __forceinline__ v16bf load_a_bf16(const bf16* p, int rs, int m, int hi) {
  const bf16* r = p + (size_t)m * rs + hi * 16;
  v8bf lo = *(const v8bf*)r;
  v8bf hb = *(const v8bf*)(r + 8);
  v16bf a;
#pragma unroll
  for (int j = 0; j < 8; ++j) { a[j] = lo[j]; a[8 + j] = hb[j]; }
  return a;
}

// Same A fragment but from fp32 memory, converted to bf16 inline.
static __device__ __forceinline__ v16bf load_a_f32(const float* p, int rs, int m, int hi) {
  const float* r = p + (size_t)m * rs + hi * 16;
  v16bf a;
#pragma unroll
  for (int j = 0; j < 8; ++j) {
    a[j]     = (bf16)r[j];
    a[8 + j] = (bf16)r[8 + j];
  }
  return a;
}

// ---------------------------------------------------------------------------
// Kernel 0: transpose + convert Wq/Wk/Wv [H,C,HS] f32 -> WT [3][H][HS][C] bf16.
// Folds the 1/sqrt(HS)=0.125 attention scale into Wq.
// ---------------------------------------------------------------------------
__global__ void k_prep_w(const float* __restrict__ Wq, const float* __restrict__ Wk,
                         const float* __restrict__ Wv, bf16* __restrict__ WT) {
  const long per = (long)NH * NC * NHS;
  long tid = (long)blockIdx.x * blockDim.x + threadIdx.x;
  if (tid >= 3 * per) return;
  int  m   = (int)(tid / per);
  long r   = tid % per;
  int  h   = (int)(r / (NC * NHS));
  int  rem = (int)(r % (NC * NHS));
  int  c   = rem / NHS;
  int  d   = rem % NHS;
  const float* W = (m == 0) ? Wq : ((m == 1) ? Wk : Wv);
  float v = W[((long)h * NC + c) * NHS + d];
  if (m == 0) v *= 0.125f;  // fold 1/sqrt(HS) into Q projection
  WT[(((long)m * NH + h) * NHS + d) * NC + c] = (bf16)v;
}

// ---------------------------------------------------------------------------
// Kernel 1: QKV projection. One wave computes a 16x64 tile (4 accumulators,
// A fragment reused across 4 WMMAs). grid = (B*S/16, H, 3 mats).
// Q,K stored [B,H,S,HS] bf16 row-major ; V stored transposed [B,H,HS,S] bf16.
// ---------------------------------------------------------------------------
__global__ void k_qkv(const float* __restrict__ x, const bf16* __restrict__ WT,
                      bf16* __restrict__ Qb, bf16* __restrict__ Kb,
                      bf16* __restrict__ Vt) {
  const int lane = threadIdx.x;
  const int n  = lane & 15;
  const int hi = lane >> 4;
  const int mt = blockIdx.x;            // token tile
  const int b  = mt / (NS / 16);
  const int st = mt % (NS / 16);
  const int h  = blockIdx.y;
  const int mat = blockIdx.z;           // 0:Q 1:K 2:V

  const float* xrow = x + ((long)b * NS + st * 16) * NC;                 // 16 x C
  const bf16*  Wt   = WT + (((long)mat * NH + h) * NHS) * NC;            // [64][C]

  v8f acc[4] = {};
  for (int k = 0; k < NC; k += 32) {
    v16bf a = load_a_f32(xrow + k, NC, n, hi);
#pragma unroll
    for (int c = 0; c < 4; ++c) {
      v16bf bb = *(const v16bf*)(Wt + (long)(c * 16 + n) * NC + k + hi * 16);
      acc[c] = wmma_bf16(a, bb, acc[c]);
    }
  }

  if (mat < 2) {
    bf16* O = (mat == 0) ? Qb : Kb;
    long base = (((long)b * NH + h) * NS + st * 16) * NHS;
#pragma unroll
    for (int c = 0; c < 4; ++c)
#pragma unroll
      for (int r = 0; r < 8; ++r) {
        int M = hi * 8 + r;
        O[base + (long)M * NHS + c * 16 + n] = (bf16)acc[c][r];
      }
  } else {
    long base = (((long)b * NH + h) * NHS) * NS + st * 16;
#pragma unroll
    for (int c = 0; c < 4; ++c)
#pragma unroll
      for (int r = 0; r < 8; ++r) {
        int M = hi * 8 + r;
        Vt[base + (long)(c * 16 + n) * NS + M] = (bf16)acc[c][r];
      }
  }
}

// ---------------------------------------------------------------------------
// Kernel 2: causal flash attention, 32 queries per wave, 32 keys per step
// (16 WMMAs / step; K/V fragments reused across both query tiles).
// grid = (S/32, B*H). Writes head-concatenated AO [B*S, C] bf16.
// ---------------------------------------------------------------------------
__global__ void k_attn(const bf16* __restrict__ Qb, const bf16* __restrict__ Kb,
                       const bf16* __restrict__ Vt, bf16* __restrict__ AO) {
  __shared__ bf16 Plds[32 * 32];
#if USE_ASYNC_LDS
  __shared__ bf16 Qlds[32 * NHS];
#endif
  const int lane = threadIdx.x;
  const int n  = lane & 15;
  const int hi = lane >> 4;
  const int qt = blockIdx.x;            // 32-query tile
  const int bh = blockIdx.y;
  const int b  = bh / NH;
  const int h  = bh % NH;
  const int q0 = qt * 32;

  const bf16* Q = Qb + ((long)bh * NS + q0) * NHS;   // 32 x 64, pre-scaled
  const bf16* K = Kb + (long)bh * NS * NHS;          // [S][HS]
  const bf16* V = Vt + (long)bh * NHS * NS;          // [HS][S]

#if USE_ASYNC_LDS
  // Stage the 4KB Q tile with ASYNCcnt-tracked global->LDS DMA (b128/lane).
#pragma unroll
  for (int i = 0; i < 8; ++i) {
    const bf16* g = Q + (i * 32 + lane) * 8;
    const bf16* l = Qlds + (i * 32 + lane) * 8;
    __builtin_amdgcn_global_load_async_to_lds_b128(as_global_v4(g), as_lds_v4(l), 0, 0);
  }
  __builtin_amdgcn_s_wait_asynccnt(0);
  __syncthreads();
  const bf16* Qsrc = Qlds;
#else
  const bf16* Qsrc = Q;
#endif

  v16bf aQ[2][2];
#pragma unroll
  for (int t = 0; t < 2; ++t)
#pragma unroll
    for (int kc = 0; kc < 2; ++kc)
      aQ[t][kc] = load_a_bf16(Qsrc + t * 16 * NHS + kc * 32, NHS, n, hi);

  float m_i[2][8], l_i[2][8];
  v8f o[2][4] = {};
#pragma unroll
  for (int t = 0; t < 2; ++t)
#pragma unroll
    for (int r = 0; r < 8; ++r) { m_i[t][r] = -1e30f; l_i[t][r] = 0.0f; }

  const int nb = qt + 1;                // 32-key blocks covering keys <= q0+31
  for (int kb = 0; kb < nb; ++kb) {
    const int base = kb * 32;
    if (kb + 1 < nb) {                  // prefetch next K/V block into caches
      __builtin_prefetch(K + (long)(base + 32 + lane) * NHS, 0, 1);
      __builtin_prefetch(V + (long)lane * NS + base + 32, 0, 1);
      __builtin_prefetch(V + (long)(32 + lane) * NS + base + 32, 0, 1);
    }
    // ---- S = Q . K^T : 8 WMMAs, K fragments shared by both q-tiles ----
    const bf16* kr0 = K + (long)(base + n) * NHS;
    const bf16* kr1 = K + (long)(base + 16 + n) * NHS;
    v16bf bk00 = *(const v16bf*)(kr0 + hi * 16);
    v16bf bk01 = *(const v16bf*)(kr0 + 32 + hi * 16);
    v16bf bk10 = *(const v16bf*)(kr1 + hi * 16);
    v16bf bk11 = *(const v16bf*)(kr1 + 32 + hi * 16);
    v8f s[2][2] = {};
#pragma unroll
    for (int t = 0; t < 2; ++t) {
      s[t][0] = wmma_bf16(aQ[t][0], bk00, s[t][0]);
      s[t][0] = wmma_bf16(aQ[t][1], bk01, s[t][0]);
      s[t][1] = wmma_bf16(aQ[t][0], bk10, s[t][1]);
      s[t][1] = wmma_bf16(aQ[t][1], bk11, s[t][1]);
    }
    // ---- causal mask + online softmax; write P straight to LDS ----
    __syncthreads();
#pragma unroll
    for (int t = 0; t < 2; ++t) {
#pragma unroll
      for (int r = 0; r < 8; ++r) {
        const int qrow = q0 + t * 16 + hi * 8 + r;
        float v0 = (base + n      > qrow) ? -1e30f : s[t][0][r];
        float v1 = (base + 16 + n > qrow) ? -1e30f : s[t][1][r];
        float mb = fmaxf(v0, v1);
#pragma unroll
        for (int off = 8; off >= 1; off >>= 1)
          mb = fmaxf(mb, __shfl_xor(mb, off, 32));   // row spans a 16-lane half
        float mn = fmaxf(m_i[t][r], mb);
        float alpha = __expf(m_i[t][r] - mn);
        float e0 = __expf(v0 - mn);
        float e1 = __expf(v1 - mn);
        float rs = e0 + e1;
#pragma unroll
        for (int off = 8; off >= 1; off >>= 1)
          rs = rs + __shfl_xor(rs, off, 32);
        l_i[t][r] = l_i[t][r] * alpha + rs;
        m_i[t][r] = mn;
        const int M = t * 16 + hi * 8 + r;
        Plds[M * 32 + n]      = (bf16)e0;
        Plds[M * 32 + 16 + n] = (bf16)e1;
#pragma unroll
        for (int c = 0; c < 4; ++c) o[t][c][r] = o[t][c][r] * alpha;
      }
    }
    __syncthreads();
    // ---- O += P . V : 8 WMMAs, V fragments shared by both q-tiles ----
    v16bf aP0 = load_a_bf16(Plds, 32, n, hi);
    v16bf aP1 = load_a_bf16(Plds + 16 * 32, 32, n, hi);
    const bf16* vb = V + base + hi * 16;
#pragma unroll
    for (int c = 0; c < 4; ++c) {
      v16bf bv = *(const v16bf*)(vb + (long)(c * 16 + n) * NS);
      o[0][c] = wmma_bf16(aP0, bv, o[0][c]);
      o[1][c] = wmma_bf16(aP1, bv, o[1][c]);
    }
    __syncthreads();
  }
  // ---- epilogue: O / l, head-concatenated layout [B*S, C] ----
  long obase = ((long)b * NS + q0) * NC + h * NHS;
#pragma unroll
  for (int t = 0; t < 2; ++t)
#pragma unroll
    for (int r = 0; r < 8; ++r) {
      int M = t * 16 + hi * 8 + r;
      float inv = 1.0f / l_i[t][r];
#pragma unroll
      for (int c = 0; c < 4; ++c)
        AO[obase + (long)M * NC + c * 16 + n] = (bf16)(o[t][c][r] * inv);
    }
}

// ---------------------------------------------------------------------------
// Kernel 3: out = AO @ Wp^T + bp. One wave per 16x64 tile (A reuse x4).
// grid = (B*S/16, C/64). B[k'][nout] = Wp[nout][k'] contiguous in k'.
// ---------------------------------------------------------------------------
__global__ void k_proj(const bf16* __restrict__ AO, const float* __restrict__ Wp,
                       const float* __restrict__ bp, float* __restrict__ out) {
  const int lane = threadIdx.x;
  const int n  = lane & 15;
  const int hi = lane >> 4;
  const int mt = blockIdx.x;
  const int nt0 = blockIdx.y * 64;

  const bf16* A = AO + (long)mt * 16 * NC;

  v8f acc[4] = {};
  for (int k = 0; k < NC; k += 32) {
    v16bf a = load_a_bf16(A + k, NC, n, hi);
#pragma unroll
    for (int c = 0; c < 4; ++c) {
      const float* p = Wp + (long)(nt0 + c * 16 + n) * NC + k + hi * 16;
      v16bf bb;
#pragma unroll
      for (int j = 0; j < 16; ++j) bb[j] = (bf16)p[j];
      acc[c] = wmma_bf16(a, bb, acc[c]);
    }
  }
#pragma unroll
  for (int c = 0; c < 4; ++c) {
    float bias = bp[nt0 + c * 16 + n];
#pragma unroll
    for (int r = 0; r < 8; ++r) {
      int M = hi * 8 + r;
      out[((long)mt * 16 + M) * NC + nt0 + c * 16 + n] = acc[c][r] + bias;
    }
  }
}

// ---------------------------------------------------------------------------
extern "C" void kernel_launch(void* const* d_in, const int* in_sizes, int n_in,
                              void* d_out, int out_size, void* d_ws, size_t ws_size,
                              hipStream_t stream) {
  const float* x  = (const float*)d_in[0];
  const float* Wq = (const float*)d_in[1];
  const float* Wk = (const float*)d_in[2];
  const float* Wv = (const float*)d_in[3];
  const float* Wp = (const float*)d_in[4];
  const float* bp = (const float*)d_in[5];
  float* out = (float*)d_out;

  char* ws = (char*)d_ws;
  size_t off = 0;
  bf16* WT = (bf16*)(ws + off); off += (size_t)3 * NH * NHS * NC * sizeof(bf16);  // 6 MB
  bf16* Qb = (bf16*)(ws + off); off += (size_t)NB * NH * NS * NHS * sizeof(bf16); // 16 MB
  bf16* Kb = (bf16*)(ws + off); off += (size_t)NB * NH * NS * NHS * sizeof(bf16); // 16 MB
  bf16* Vt = (bf16*)(ws + off); off += (size_t)NB * NH * NS * NHS * sizeof(bf16); // 16 MB
  bf16* AO = (bf16*)(ws + off); off += (size_t)NB * NS * NC * sizeof(bf16);       // 16 MB

  {
    long total = (long)3 * NH * NC * NHS;
    k_prep_w<<<dim3((unsigned)((total + 255) / 256)), 256, 0, stream>>>(Wq, Wk, Wv, WT);
  }
  k_qkv <<<dim3(NB * NS / 16, NH, 3),  32, 0, stream>>>(x, WT, Qb, Kb, Vt);
  k_attn<<<dim3(NS / 32, NB * NH),     32, 0, stream>>>(Qb, Kb, Vt, AO);
  k_proj<<<dim3(NB * NS / 16, NC / 64),32, 0, stream>>>(AO, Wp, bp, out);
}
